// SimpleGPT_79370995630810
// MI455X (gfx1250) — matrix operand order repaired and checked by reading, not compiled, not run on hardware
//
#include <hip/hip_runtime.h>
#include <hip/hip_bf16.h>
#include <math.h>

// ---------------------------------------------------------------------------
// SimpleGPT forward on MI455X (gfx1250, wave32, WMMA).
// Residual stream fp32; GEMM operands bf16 with f32 WMMA accumulation.
// All B matrices are pre-packed into WMMA fragment-major layout so the GEMM
// inner loop is pure b128 loads + v_wmma.
// ---------------------------------------------------------------------------

typedef __bf16 bf16_t;
typedef __attribute__((ext_vector_type(16))) __bf16 v16bf;
typedef __attribute__((ext_vector_type(8)))  __bf16 v8bf;
typedef __attribute__((ext_vector_type(8)))  float  v8f;

#define GB 32
#define GT 80
#define GD 128
#define GH 8
#define GDH 64
#define GDMLP 512
#define GL 8
#define GV 50257
#define GM (GB * GT)            // 2560 token rows

// ---------------------------------------------------------------------------
// CDNA5 16-bit WMMA lane layouts (CDNA5 ISA):
//   A (16x32, MxK): lane L -> row m = L%16 ; element e -> K:
//       e<8 : K = (L<16 ? 0 : 8)  + e      (contiguous run of 8)
//       e>=8: K = (L<16 ? 16: 24) + (e-8)  (contiguous run of 8)
//   B (32x16, KxN): lane L -> col n = L%16 ; same K order as A; code loads A and B with K = (L<16?0:16)+e
//   C/D (16x16 f32): lane L -> col n = L%16 ; VGPR r -> row m = r + (L<16?0:8)
//
// Packed-B layout: for each (ntile, ktile) a 512-element block at
//   ((nt*KT)+kt)*512 + lane*16 + e   holding B[kt*32+(L<16?0:16)+e, nt*16+L%16]
// -> each lane's fragment is 32 contiguous bytes (two b128 loads).
// ---------------------------------------------------------------------------

__device__ inline v16bf load_a_frag(const bf16_t* __restrict__ rowp, bool hi) {
    v8bf lo = *(const v8bf*)(rowp + (hi ? 16 : 0));
    v8bf h8 = *(const v8bf*)(rowp + (hi ? 24 : 8));
    v16bf f;
#pragma unroll
    for (int e = 0; e < 8; ++e) { f[e] = lo[e]; f[e + 8] = h8[e]; }
    return f;
}

__device__ inline v16bf load_contig16(const bf16_t* __restrict__ p) {
    v8bf lo = *(const v8bf*)p;
    v8bf h8 = *(const v8bf*)(p + 8);
    v16bf f;
#pragma unroll
    for (int e = 0; e < 8; ++e) { f[e] = lo[e]; f[e + 8] = h8[e]; }
    return f;
}

// ---------------------------------------------------------------------------
// Weight packing kernels (one-time per launch, deterministic)
// ---------------------------------------------------------------------------

// Pack a row-major f32 [K,N] matrix into fragment-major bf16 blocks.
__global__ void pack_b_f32(const float* __restrict__ src, bf16_t* __restrict__ dst,
                           int K, int N, int total) {
    int i = blockIdx.x * blockDim.x + threadIdx.x;
    if (i >= total) return;
    int w    = i & 511;
    int blk  = i >> 9;
    int lane = w >> 4, e = w & 15;
    int KT = K >> 5;
    int nt = blk / KT, kt = blk % KT;
    int k   = kt * 32 + ((lane & 16) ? 16 : 0) + e;
    int col = nt * 16 + (lane & 15);
    dst[i] = (col < N) ? (bf16_t)src[(size_t)k * N + col] : (bf16_t)0.f;
}

// Pack one layer's fused QKV weight [D=128, 1536] (virtual, from (H,D,DH)
// fp32 per-head weights) directly into fragment-major blocks. KT=4, NT=96.
__global__ void pack_qkv(const float* __restrict__ Wq, const float* __restrict__ Wk,
                         const float* __restrict__ Wv, bf16_t* __restrict__ dst) {
    int i = blockIdx.x * blockDim.x + threadIdx.x;   // total = 4*96*512 = 196608
    int w    = i & 511;
    int blk  = i >> 9;
    int lane = w >> 4, e = w & 15;
    int nt = blk >> 2, kt = blk & 3;                 // KT = 4
    int k   = kt * 32 + ((lane & 16) ? 16 : 0) + e;  // k in [0,128)
    int col = nt * 16 + (lane & 15);                 // col in [0,1536)
    int sel = col >> 9, rem = col & 511;
    int hh  = rem >> 6, e2 = rem & 63;
    const float* W = (sel == 0) ? Wq : ((sel == 1) ? Wk : Wv);
    dst[i] = (bf16_t)W[((size_t)hh * GD + k) * GDH + e2];
}

// ---------------------------------------------------------------------------
// Embedding: h[m,d] = tok_emb[x[m],d] + pos_emb[m%T,d]
// ---------------------------------------------------------------------------

__global__ void embed_kernel(const int* __restrict__ x, const float* __restrict__ tok,
                             const float* __restrict__ pos, float* __restrict__ h) {
    int m = blockIdx.x, d = threadIdx.x;
    h[(size_t)m * GD + d] = tok[(size_t)x[m] * GD + d] + pos[(size_t)(m % GT) * GD + d];
}

// ---------------------------------------------------------------------------
// LayerNorm over D=128, wave32 per row (4 f32/lane + shfl_xor reduce) -> bf16
// ---------------------------------------------------------------------------

__global__ void layernorm_kernel(const float* __restrict__ h, const float* __restrict__ g,
                                 const float* __restrict__ b, bf16_t* __restrict__ out, int M) {
    int wave = threadIdx.x >> 5, lane = threadIdx.x & 31;
    int row = blockIdx.x * 8 + wave;
    if (row >= M) return;
    const float* p = h + (size_t)row * GD + lane * 4;
    float4 v = *(const float4*)p;
    float s  = v.x + v.y + v.z + v.w;
    float sq = v.x * v.x + v.y * v.y + v.z * v.z + v.w * v.w;
#pragma unroll
    for (int o = 16; o > 0; o >>= 1) {
        s  += __shfl_xor(s,  o, 32);
        sq += __shfl_xor(sq, o, 32);
    }
    float mean = s * (1.0f / GD);
    float var  = sq * (1.0f / GD) - mean * mean;   // population var (jnp.var)
    float r    = rsqrtf(var + 1e-5f);
    const float* vp = &v.x;
#pragma unroll
    for (int j = 0; j < 4; ++j) {
        int d = lane * 4 + j;
        out[(size_t)row * GD + d] = (bf16_t)((vp[j] - mean) * r * g[d] + b[d]);
    }
}

// ---------------------------------------------------------------------------
// Generic GEMM: C[M,N] = act(A[M,K](bf16) @ Bpacked + bias) (+ Res)
// One wave computes a 16x64 output stripe (4 accumulators): the A fragment is
// reused across 4 WMMAs; all operand loads are contiguous b128.
// K mult of 32, M mult of 16, NTpad (n-tiles incl. padding) mult of 4.
// ---------------------------------------------------------------------------

template <bool OUT_BF16, bool BIAS, bool GELU, bool RES>
__global__ void gemm_wmma4(const bf16_t* __restrict__ A, const bf16_t* __restrict__ Bp,
                           const float* __restrict__ bias, const float* __restrict__ Res,
                           void* __restrict__ Cout, int M, int N, int K, int NTpad) {
    int KT = K >> 5;
    int NG = NTpad >> 2;
    int total = (M >> 4) * NG;
    int wv = blockIdx.x * (blockDim.x >> 5) + (threadIdx.x >> 5);
    if (wv >= total) return;                 // wave-uniform (EXEC all-1 for WMMA)
    int tm = wv / NG, g = wv % NG;
    int lane = threadIdx.x & 31;
    int lm = lane & 15;
    bool hi = lane >= 16;
    int rowA = tm * 16 + lm;

    v8f acc[4];
#pragma unroll
    for (int j = 0; j < 4; ++j) acc[j] = (v8f){0.f, 0.f, 0.f, 0.f, 0.f, 0.f, 0.f, 0.f};

    const bf16_t* Ar    = A + (size_t)rowA * K;
    const bf16_t* Bbase = Bp + ((size_t)g * 4 * KT) * 512 + lane * 16;
    for (int kt = 0; kt < KT; ++kt) {
        v16bf fa = load_a_frag(Ar + kt * 32, hi);
#pragma unroll
        for (int j = 0; j < 4; ++j) {
            v16bf fb = load_contig16(Bbase + ((size_t)j * KT + kt) * 512);
            acc[j] = __builtin_amdgcn_wmma_f32_16x16x32_bf16(false, fa, false, fb,
                                                             (short)0, acc[j], false, false);
        }
    }

#pragma unroll
    for (int j = 0; j < 4; ++j) {
        int col = (g * 4 + j) * 16 + lm;
        if (col >= N) continue;              // ragged classifier edge
        float bv = BIAS ? bias[col] : 0.f;
#pragma unroll
        for (int r = 0; r < 8; ++r) {
            int m = tm * 16 + r + (hi ? 8 : 0);
            float v = acc[j][r] + bv;
            if (GELU) v = 0.5f * v * (1.f + erff(v * 0.70710678118654752f)); // exact GELU
            size_t idx = (size_t)m * N + col;
            if (RES) v += Res[idx];
            if (OUT_BF16) ((bf16_t*)Cout)[idx] = (bf16_t)v;
            else          ((float*)Cout)[idx]  = v;
        }
    }
}

// ---------------------------------------------------------------------------
// Attention: one workgroup (5 waves / 160 thr) per (b,h).
//  LDS: Qs/Ks [80][64] bf16 (20480B) + Vt [64][96] bf16 transposed+padded
//  (12288B) + Sc [80][80] f32 (25600B) = 58368B < 64KB.
//  P (bf16, row stride 160, 96 cols zero-padded) is aliased over Sc: within
//  each row t the softmax reads Sc[t][s] (byte 4s) strictly before writing
//  P[t][s] (byte 2s) and row bases coincide, so the overlay is safe.
// ---------------------------------------------------------------------------

__global__ void attention_kernel(const bf16_t* __restrict__ qkv, bf16_t* __restrict__ obuf) {
    __shared__ bf16_t Qs[GT * GDH];
    __shared__ bf16_t Ks[GT * GDH];
    __shared__ bf16_t Vt[GDH * 96];               // transposed: Vt[e][t], t padded to 96
    __shared__ float  Sc[GT * GT];
    bf16_t* Pb = (bf16_t*)Sc;                     // row t at Pb + t*160, 96 cols

    int bh = blockIdx.x;
    int b = bh >> 3, h = bh & 7;
    int tid = threadIdx.x;
    const bf16_t* base = qkv + (size_t)b * GT * 1536 + h * GDH;

    // Stage Q/K (row-major) and V (transposed) with b128 global loads
    for (int i = tid; i < GT * 8; i += 160) {     // 640 chunks of 8 elements
        int t = i >> 3, ec = (i & 7) * 8;
        size_t gsrc = (size_t)t * 1536 + ec;
        *(v8bf*)(Qs + t * GDH + ec) = *(const v8bf*)(base + gsrc);
        *(v8bf*)(Ks + t * GDH + ec) = *(const v8bf*)(base + gsrc + 512);
        v8bf vv = *(const v8bf*)(base + gsrc + 1024);
#pragma unroll
        for (int j = 0; j < 8; ++j) Vt[(ec + j) * 96 + t] = vv[j];
    }
    for (int i = tid; i < GDH * 16; i += 160) {   // zero-pad Vt rows t=80..95
        int e = i >> 4, t = GT + (i & 15);
        Vt[e * 96 + t] = (bf16_t)0.f;
    }
    __syncthreads();

    int w = tid >> 5, lane = tid & 31, lm = lane & 15;
    bool hi = lane >= 16;

    // ---- Stage 1: raw scores = Q K^T (causal tile skipping: s-tile <= t-tile)
    for (int st = 0; st <= w; ++st) {
        v8f acc = {0.f, 0.f, 0.f, 0.f, 0.f, 0.f, 0.f, 0.f};
#pragma unroll
        for (int kb = 0; kb < GDH; kb += 32) {
            v16bf fa = load_a_frag(Qs + (size_t)(w * 16 + lm) * GDH + kb, hi);
            // B(k,n) = K[s=n][k] -> contiguous in k within a row of Ks
            v16bf fb = load_contig16(Ks + (size_t)(st * 16 + lm) * GDH + kb + (hi ? 16 : 0));
            acc = __builtin_amdgcn_wmma_f32_16x16x32_bf16(false, fa, false, fb,
                                                          (short)0, acc, false, false);
        }
#pragma unroll
        for (int r = 0; r < 8; ++r) {
            int t = w * 16 + r + (hi ? 8 : 0);
            Sc[t * GT + st * 16 + lm] = acc[r];
        }
    }
    __syncthreads();

    // ---- Softmax with faithful masked_fill(tril(scores)==0, -inf)
    if (tid < GT) {
        int t = tid;
        const float scale = 0.125f;               // 1/sqrt(64)
        float mx = -__builtin_inff();
        for (int s = 0; s <= t; ++s) {
            float x = Sc[t * GT + s] * scale;
            if (x != 0.f) mx = fmaxf(mx, x);      // x==0 entries are masked too
        }
        float sum = 0.f;
        for (int s = 0; s <= t; ++s) {
            float x = Sc[t * GT + s] * scale;
            sum += (x == 0.f) ? 0.f : expf(x - mx);
        }
        float inv = 1.f / sum;
        for (int s = 0; s < 96; ++s) {            // read Sc[t][s] before writing Pb[t][s]
            float p = 0.f;
            if (s <= t) {
                float x = Sc[t * GT + s] * scale;
                p = (x == 0.f) ? 0.f : expf(x - mx) * inv;
            }
            Pb[t * 160 + s] = (bf16_t)p;          // overlays dead Sc bytes (safe ordering)
        }
    }
    __syncthreads();

    // ---- Stage 2: O = P @ V  (K padded 80 -> 96 with zeros; all b128 LDS loads)
#pragma unroll
    for (int et = 0; et < 4; ++et) {
        v8f acc = {0.f, 0.f, 0.f, 0.f, 0.f, 0.f, 0.f, 0.f};
#pragma unroll
        for (int kb = 0; kb < 96; kb += 32) {
            v16bf fa = load_a_frag(Pb + (size_t)(w * 16 + lm) * 160 + kb, hi);
            // B(k=t, n=e) = Vt[e][t] -> contiguous in k within a row of Vt
            v16bf fb = load_contig16(Vt + (size_t)(et * 16 + lm) * 96 + kb + (hi ? 16 : 0));
            acc = __builtin_amdgcn_wmma_f32_16x16x32_bf16(false, fa, false, fb,
                                                          (short)0, acc, false, false);
        }
#pragma unroll
        for (int r = 0; r < 8; ++r) {
            int t = w * 16 + r + (hi ? 8 : 0);
            // o layout [B,T,H*DH] to feed the Wo GEMM directly
            obuf[((size_t)b * GT + t) * (GH * GDH) + h * GDH + et * 16 + lm] = (bf16_t)acc[r];
        }
    }
}

// ---------------------------------------------------------------------------
// Host-side orchestration (graph-capture safe: launches only, all on stream)
// ---------------------------------------------------------------------------

static inline int gemm_blocks4(int M, int NTpad) {
    int waves = (M >> 4) * (NTpad >> 2);
    return (waves + 7) / 8;                       // 8 waves (256 thr) per block
}

extern "C" void kernel_launch(void* const* d_in, const int* in_sizes, int n_in,
                              void* d_out, int out_size, void* d_ws, size_t ws_size,
                              hipStream_t stream) {
    (void)in_sizes; (void)n_in; (void)out_size; (void)ws_size;
    const int*   x     = (const int*)  d_in[0];
    const float* tok   = (const float*)d_in[1];
    const float* pos   = (const float*)d_in[2];
    const float* Wq    = (const float*)d_in[3];
    const float* Wk    = (const float*)d_in[4];
    const float* Wv    = (const float*)d_in[5];
    const float* Wo    = (const float*)d_in[6];
    const float* bo    = (const float*)d_in[7];
    const float* ln1g  = (const float*)d_in[8];
    const float* ln1b  = (const float*)d_in[9];
    const float* ln2g  = (const float*)d_in[10];
    const float* ln2b  = (const float*)d_in[11];
    const float* W1    = (const float*)d_in[12];
    const float* b1    = (const float*)d_in[13];
    const float* W2    = (const float*)d_in[14];
    const float* b2    = (const float*)d_in[15];
    const float* lnfg  = (const float*)d_in[16];
    const float* lnfb  = (const float*)d_in[17];
    const float* Wcls  = (const float*)d_in[18];
    const float* bcls  = (const float*)d_in[19];
    float* out = (float*)d_out;

    // Packed sizes (elements): blocks of 512 bf16 per (ktile,ntile)
    const size_t PQKV = 4  * 96   * 512;   // per layer: K=128  N=1536
    const size_t PWO  = 16 * 8    * 512;   // per layer: K=512  N=128
    const size_t PW1  = 4  * 32   * 512;   // per layer: K=128  N=512
    const size_t PW2  = 16 * 8    * 512;   // per layer: K=512  N=128
    const size_t PCLS = 4  * 3144 * 512;   // K=128 N=50257 (NT padded 3142->3144)

    // Workspace carve-up (256B aligned slices)
    char* W = (char*)d_ws;
    size_t off = 0;
    auto take = [&](size_t bytes) { size_t r = off; off += (bytes + 255) & ~(size_t)255; return r; };
    bf16_t* wqkv = (bf16_t*)(W + take(GL * PQKV * 2));
    bf16_t* wo   = (bf16_t*)(W + take(GL * PWO  * 2));
    bf16_t* w1   = (bf16_t*)(W + take(GL * PW1  * 2));
    bf16_t* w2   = (bf16_t*)(W + take(GL * PW2  * 2));
    bf16_t* wcls = (bf16_t*)(W + take(PCLS * 2));
    float*  hbuf = (float*) (W + take((size_t)GM * GD * 4));
    bf16_t* abuf = (bf16_t*)(W + take((size_t)GM * GD * 2));
    bf16_t* qkvb = (bf16_t*)(W + take((size_t)GM * 1536 * 2));
    bf16_t* obuf = (bf16_t*)(W + take((size_t)GM * 512 * 2));
    bf16_t* midb = (bf16_t*)(W + take((size_t)GM * GDMLP * 2));

    // One-time weight packing into fragment-major bf16
    for (int l = 0; l < GL; ++l) {
        pack_qkv<<<(int)(PQKV / 256), 256, 0, stream>>>(
            Wq + (size_t)l * GH * GD * GDH, Wk + (size_t)l * GH * GD * GDH,
            Wv + (size_t)l * GH * GD * GDH, wqkv + l * PQKV);
        pack_b_f32<<<(int)(PWO / 256), 256, 0, stream>>>(
            Wo + (size_t)l * 512 * GD, wo + l * PWO, 512, GD, (int)PWO);
        pack_b_f32<<<(int)(PW1 / 256), 256, 0, stream>>>(
            W1 + (size_t)l * GD * GDMLP, w1 + l * PW1, GD, GDMLP, (int)PW1);
        pack_b_f32<<<(int)(PW2 / 256), 256, 0, stream>>>(
            W2 + (size_t)l * GDMLP * GD, w2 + l * PW2, GDMLP, GD, (int)PW2);
    }
    pack_b_f32<<<(int)((PCLS + 255) / 256), 256, 0, stream>>>(
        Wcls, wcls, GD, GV, (int)PCLS);

    // Embedding
    embed_kernel<<<GM, GD, 0, stream>>>(x, tok, pos, hbuf);

    // Transformer blocks
    for (int l = 0; l < GL; ++l) {
        layernorm_kernel<<<GM / 8, 256, 0, stream>>>(hbuf, ln1g + l * GD, ln1b + l * GD, abuf, GM);
        // fused QKV: [2560,128] @ [128,1536] -> bf16
        gemm_wmma4<true, false, false, false><<<gemm_blocks4(GM, 96), 256, 0, stream>>>(
            abuf, wqkv + l * PQKV, nullptr, nullptr, qkvb, GM, 1536, GD, 96);
        attention_kernel<<<GB * GH, 160, 0, stream>>>(qkvb, obuf);
        // h = h + o @ Wo + bo
        gemm_wmma4<false, true, false, true><<<gemm_blocks4(GM, 8), 256, 0, stream>>>(
            obuf, wo + l * PWO, bo + l * GD, hbuf, hbuf, GM, GD, 512, 8);
        layernorm_kernel<<<GM / 8, 256, 0, stream>>>(hbuf, ln2g + l * GD, ln2b + l * GD, abuf, GM);
        // mid = gelu(a @ W1 + b1) -> bf16
        gemm_wmma4<true, true, true, false><<<gemm_blocks4(GM, 32), 256, 0, stream>>>(
            abuf, w1 + l * PW1, b1 + l * GDMLP, nullptr, midb, GM, GDMLP, GD, 32);
        // h = h + mid @ W2 + b2
        gemm_wmma4<false, true, false, true><<<gemm_blocks4(GM, 8), 256, 0, stream>>>(
            midb, w2 + l * PW2, b2 + l * GD, hbuf, hbuf, GM, GD, GDMLP, 8);
    }

    // Final LN + classifier (dominant: 33 GFLOP, 515 MB fp32 writeback)
    layernorm_kernel<<<GM / 8, 256, 0, stream>>>(hbuf, lnfg, lnfb, abuf, GM);
    gemm_wmma4<false, true, false, false><<<gemm_blocks4(GM, 3144), 256, 0, stream>>>(
        abuf, wcls, bcls, nullptr, out, GM, GV, GD, 3144);
}
